// BiasedScanAttention_82824149336916
// MI455X (gfx1250) — matrix operand order repaired and checked by reading, not compiled, not run on hardware
//
#include <hip/hip_runtime.h>

typedef __attribute__((ext_vector_type(16))) _Float16 v16h;
typedef __attribute__((ext_vector_type(8)))  _Float16 v8h;
typedef __attribute__((ext_vector_type(8)))  float    v8f;

#define B_   2
#define H_   8
#define Q_   2048
#define K_   2048
#define D_   64
#define DV_  64
#define DS_  3
#define M_   8
#define KT   32            // keys per K-tile iteration
#define NW   8             // waves per block == heads
#define QT   16            // q rows per block (shared by all heads)
#define NIT  (K_/KT)

// xor-shuffle max within a 16-lane DPP row (VALU pipe, no LDS traffic)
template<int XMASK>
__device__ __forceinline__ float xmax16(float x) {
    const int y = __builtin_amdgcn_update_dpp(
        __float_as_int(x), __float_as_int(x), 0x160 | XMASK, 0xf, 0xf, true);
    return fmaxf(x, __int_as_float(y));
}

__global__ __launch_bounds__(256) void biased_flash_attn_kernel(
    const float* __restrict__ qs,  const float* __restrict__ ks,
    const float* __restrict__ vs,  const float* __restrict__ qssg,
    const float* __restrict__ kssg,const float* __restrict__ ls,
    const float* __restrict__ rw,  const int*   __restrict__ mask,
    float* __restrict__ out)
{
    __shared__ __align__(32) _Float16 sK  [NW][KT][D_];    // 32 KB per-head K tile [key][d]
    __shared__ __align__(32) _Float16 sVt [NW][DV_][KT];   // 32 KB per-head V tile [d][key]
    __shared__ __align__(32) _Float16 sP  [NW][QT][KT];    //  8 KB per-wave P staging
    __shared__ float sBias[2][NW][QT][KT];                 // 32 KB double-buffered bias (mask folded)
    __shared__ float sQss [QT][4];

    const int tid  = threadIdx.x;
    const int wid  = tid >> 5;          // wave id == head id
    const int lane = tid & 31;
    const int lid  = lane & 15;
    const int half = lane >> 4;

    const int nqb = Q_ / QT;
    const int b   = blockIdx.x / nqb;
    const int qb  = (blockIdx.x % nqb) * QT;
    const size_t bh = (size_t)b * H_ + wid;

    float gamma[M_];
#pragma unroll
    for (int m = 0; m < M_; ++m) { const float l = ls[m]; gamma[m] = 1.0f/(2.0f*l*l); }

    if (tid < QT) {
        const float* p = qssg + ((size_t)b*Q_ + qb + tid)*DS_;
        sQss[tid][0]=p[0]; sQss[tid][1]=p[1]; sQss[tid][2]=p[2]; sQss[tid][3]=0.f;
    }

    // --- Q A-fragments (16x32 f16 per d-chunk) ---
    const float* qp = qs + (bh*Q_ + qb + lid)*D_;
    v16h aq[2];
#pragma unroll
    for (int c = 0; c < 2; ++c) {
        const int dbase = 32*c + half*16;
#pragma unroll
        for (int e = 0; e < 8; ++e) {
            aq[c][e]   = (_Float16)qp[dbase + e];
            aq[c][e+8] = (_Float16)qp[dbase + 8 + e];
        }
    }

    // all-ones B operand: P x ones -> per-row sum of probabilities in every lane
    v16h vones;
#pragma unroll
    for (int e = 0; e < 16; ++e) vones[e] = (_Float16)1.0f;

    v8f o[4], o4;                       // o4 = row-sum accumulator
    float rmax_s = -1e30f;              // group-shared running max (uniform in 16-lane group)
#pragma unroll
    for (int n = 0; n < 4; ++n) o[n] = (v8f){0,0,0,0,0,0,0,0};
    o4 = (v8f){0,0,0,0,0,0,0,0};

    // cooperative-phase mapping: thread owns key `ck` and q rows {cq2, cq2+1}
    const int ck  = tid & 31;
    const int cq2 = (tid >> 5) * 2;

    auto coop_bias = [&](int k0c, int buf) {
        const float* kp = kssg + ((size_t)b*K_ + k0c + ck)*DS_;
        const float ks0 = kp[0], ks1 = kp[1], ks2 = kp[2];
        const float mb  = mask[(size_t)b*K_ + k0c + ck] ? 0.0f : -1e30f;
#pragma unroll
        for (int r = 0; r < 2; ++r) {
            const int q = cq2 + r;
            const float d0 = sQss[q][0]-ks0, d1 = sQss[q][1]-ks1, d2v = sQss[q][2]-ks2;
            const float dd = d0*d0 + d1*d1 + d2v*d2v;
            float phi[M_];
#pragma unroll
            for (int m = 0; m < M_; ++m) phi[m] = __expf(-dd * gamma[m]);
#pragma unroll
            for (int hh = 0; hh < H_; ++hh) {
                float bias = mb;
#pragma unroll
                for (int m = 0; m < M_; ++m) bias += rw[hh*M_ + m] * phi[m];
                sBias[buf][hh][q][ck] = bias;
            }
        }
    };

    __syncthreads();        // sQss visible
    coop_bias(0, 0);        // prologue: bias tile for kt = 0

    for (int kt = 0; kt < NIT; ++kt) {
        const int k0 = kt*KT;
        __syncthreads();    // bias[kt&1] ready; prev reads of bias[(kt+1)&1] done

        // ---- per-wave K tile: lane = key, fp32 -> f16 ----
        {
            const float* kp = ks + (bh*K_ + k0 + lane)*D_;
#pragma unroll
            for (int d = 0; d < D_; d += 8) {
                const float4 f0 = *(const float4*)(kp + d);
                const float4 f1 = *(const float4*)(kp + d + 4);
                v8h t;
                t[0]=(_Float16)f0.x; t[1]=(_Float16)f0.y; t[2]=(_Float16)f0.z; t[3]=(_Float16)f0.w;
                t[4]=(_Float16)f1.x; t[5]=(_Float16)f1.y; t[6]=(_Float16)f1.z; t[7]=(_Float16)f1.w;
                *(v8h*)&sK[wid][lane][d] = t;
            }
        }
        // ---- per-wave V tile, transposed ----
        {
            const int kp2 = (lane & 15)*2;
            const int dh  = (lane >> 4)*32;
            const float* v0 = vs + (bh*K_ + k0 + kp2)*DV_ + dh;
            const float* v1 = v0 + DV_;
#pragma unroll
            for (int d = 0; d < 32; d += 4) {
                const float4 x = *(const float4*)(v0 + d);
                const float4 y = *(const float4*)(v1 + d);
                _Float16* p0 = &sVt[wid][dh+d+0][kp2]; p0[0]=(_Float16)x.x; p0[1]=(_Float16)y.x;
                _Float16* p1 = &sVt[wid][dh+d+1][kp2]; p1[0]=(_Float16)x.y; p1[1]=(_Float16)y.y;
                _Float16* p2 = &sVt[wid][dh+d+2][kp2]; p2[0]=(_Float16)x.z; p2[1]=(_Float16)y.z;
                _Float16* p3 = &sVt[wid][dh+d+3][kp2]; p3[0]=(_Float16)x.w; p3[1]=(_Float16)y.w;
            }
        }

        // ---- cooperative bias for NEXT tile (other buffer) ----
        if (kt + 1 < NIT) coop_bias(k0 + KT, (kt + 1) & 1);

        // ---- S = Q * K^T ----
        v8f s[2];
#pragma unroll
        for (int t = 0; t < 2; ++t) {
            v8f acc = (v8f){0,0,0,0,0,0,0,0};
#pragma unroll
            for (int c = 0; c < 2; ++c) {
                const v16h bk = *(const v16h*)&sK[wid][t*16 + lid][32*c + half*16];
                acc = __builtin_amdgcn_wmma_f32_16x16x32_f16(
                          false, aq[c], false, bk, (short)0, acc, false, false);
            }
            s[t] = acc;
        }

        // ---- bias add + group-max online softmax ----
        const int bufc = kt & 1;
        float tm = -1e30f;
#pragma unroll
        for (int j = 0; j < 8; ++j) {
            const int row = j + half*8;
#pragma unroll
            for (int t = 0; t < 2; ++t) {
                s[t][j] = s[t][j]*0.125f + sBias[bufc][wid][row][t*16 + lid];
                tm = fmaxf(tm, s[t][j]);
            }
        }
        // one xor tree for the whole half-tile (softmax is shift-invariant;
        // mask is per-column so all rows in the group share its pattern)
        tm = xmax16<8>(tm); tm = xmax16<4>(tm); tm = xmax16<2>(tm); tm = xmax16<1>(tm);
        const float mnew  = fmaxf(rmax_s, tm);
        const float scale = __expf(rmax_s - mnew);
        rmax_s = mnew;
#pragma unroll
        for (int n = 0; n < 4; ++n) o[n] = o[n] * scale;
        o4 = o4 * scale;
#pragma unroll
        for (int j = 0; j < 8; ++j) {
            const int row = j + half*8;
            const float p0 = __expf(s[0][j] - mnew);
            const float p1 = __expf(s[1][j] - mnew);
            sP[wid][row][lid]      = (_Float16)p0;
            sP[wid][row][16 + lid] = (_Float16)p1;
        }

        // ---- relayout P: C layout -> A layout via per-wave LDS ----
        v16h ap;
        {
            const _Float16* pp = &sP[wid][lid][half*16];
            const v8h lo = *(const v8h*)(pp);
            const v8h hi = *(const v8h*)(pp + 8);
#pragma unroll
            for (int e = 0; e < 8; ++e) { ap[e] = lo[e]; ap[e+8] = hi[e]; }
        }

        // ---- O += P * V (4 chunks) and row-sum += P * ones (1 extra WMMA) ----
#pragma unroll
        for (int n = 0; n < 4; ++n) {
            const v16h bv = *(const v16h*)&sVt[wid][n*16 + lid][half*16];
            o[n] = __builtin_amdgcn_wmma_f32_16x16x32_f16(
                       false, ap, false, bv, (short)0, o[n], false, false);
        }
        o4 = __builtin_amdgcn_wmma_f32_16x16x32_f16(
                 false, ap, false, vones, (short)0, o4, false, false);
    }

    // ---- epilogue: normalize and store fp32 (o4[j] = exact row sum in every lane) ----
#pragma unroll
    for (int j = 0; j < 8; ++j) {
        const float inv = 1.0f / (o4[j] + 1e-10f);
        const int qr = qb + j + half*8;
        float* op = out + (bh*Q_ + qr)*DV_;
#pragma unroll
        for (int n = 0; n < 4; ++n) op[n*16 + lid] = o[n][j] * inv;
    }
}

extern "C" void kernel_launch(void* const* d_in, const int* in_sizes, int n_in,
                              void* d_out, int out_size, void* d_ws, size_t ws_size,
                              hipStream_t stream) {
    const float* qs  = (const float*)d_in[0];
    const float* ks  = (const float*)d_in[1];
    const float* vs  = (const float*)d_in[2];
    const float* qss = (const float*)d_in[3];
    const float* kss = (const float*)d_in[4];
    const float* ls  = (const float*)d_in[5];
    const float* rw  = (const float*)d_in[6];
    const int*   msk = (const int*)d_in[7];
    float* out = (float*)d_out;

    const dim3 grid(B_ * (Q_ / QT));   // 256 blocks x 256 threads (8 waves = 8 heads)
    biased_flash_attn_kernel<<<grid, 256, 0, stream>>>(qs, ks, vs, qss, kss, ls, rw, msk, out);
    (void)in_sizes; (void)n_in; (void)out_size; (void)d_ws; (void)ws_size;
}
